// LSTMRegressionModel_17145509446512
// MI455X (gfx1250) — matrix operand order repaired and checked by reading, not compiled, not run on hardware
//
#include <hip/hip_runtime.h>

// ---------------------------------------------------------------------------
// LSTM on MI455X (gfx1250): bf16 WMMA (v_wmma_f32_16x16x32_bf16), f32 accum.
//   xg = x @ W_ih^T + b_ih          (one 34-GFLOP GEMM, done once)
//   15x: gates = xg + h @ W_hh^T + b_hh ; c,h update ; store h slice
// W_hh (8MB bf16) + xg (32MB f32) + h (4MB bf16) stay L2-resident (192MB L2);
// HBM traffic ~190MB total -> recurrence is compute/L2 bound, so we use the
// deepest 16-bit WMMA (K=32) instead of the shallow K=4 f32 WMMA.
// ---------------------------------------------------------------------------

typedef __attribute__((ext_vector_type(16))) __bf16          v16bf;
typedef __attribute__((ext_vector_type(8)))  float           v8f;
typedef __attribute__((ext_vector_type(8)))  unsigned short  v8us;

union FragU { v16bf v; v8us h[2]; };

static __device__ __forceinline__ unsigned short f32_to_bf16(float f) {
  union { float f; unsigned int u; } c; c.f = f;
  unsigned int u = c.u;
  unsigned int r = u + 0x7FFFu + ((u >> 16) & 1u);   // round-to-nearest-even
  return (unsigned short)(r >> 16);
}

static __device__ __forceinline__ float fast_sigmoid(float x) {
  return 1.0f / (1.0f + __builtin_amdgcn_exp2f(-1.44269504f * x));
}
static __device__ __forceinline__ float fast_tanh(float x) {
  float e = __builtin_amdgcn_exp2f(-2.88539008f * x);  // exp(-2x)
  return (1.0f - e) / (1.0f + e);
}

// A fragment: 16x32 bf16, rows m0..m0+15, K = k0..k0+31, row-major src.
// Layout: lanes 0-15 row M=lane, K 0..15 contiguous;
//         lanes 16-31 same rows, K 16..31 (matches B fragment order).
static __device__ __forceinline__ v16bf
load_a_frag(const unsigned short* __restrict__ A, int lda, int m0, int k0, int lane) {
  const int row = m0 + (lane & 15);
  const int kb  = k0 + ((lane & 16) ? 16 : 0);
  const unsigned short* p = A + (size_t)row * lda + kb;
  FragU f;
  f.h[0] = *reinterpret_cast<const v8us*>(p);        // K kb..kb+7   -> VGPR0-3
  f.h[1] = *reinterpret_cast<const v8us*>(p + 8);    // K kb+8..+15  -> VGPR4-7
  return f.v;
}

// B fragment: 32x16 bf16 = (W^T) tile; B[k][n] = W[n][k], W row-major.
// Layout used here (contiguous k, same as A): lane n%16 holds column n; lanes 0-15 K=0..15, lanes 16-31 K=16..31,
// packed 2/VGPR ascending -> one contiguous 32B load per lane.
static __device__ __forceinline__ v16bf
load_bT_frag(const unsigned short* __restrict__ W, int ldw, int n0, int k0, int lane) {
  const int row = n0 + (lane & 15);
  const int kb  = k0 + ((lane & 16) ? 16 : 0);
  const unsigned short* p = W + (size_t)row * ldw + kb;
  FragU f;
  f.h[0] = *reinterpret_cast<const v8us*>(p);
  f.h[1] = *reinterpret_cast<const v8us*>(p + 8);
  return f.v;
}

#define WMMA_BF16(a, b, acc) \
  __builtin_amdgcn_wmma_f32_16x16x32_bf16(false, (a), false, (b), (short)0, (acc), false, false)

// ---------------------------------------------------------------------------
// xg = x @ W_ih^T + b_ih    x:[2048,2048] W_ih:[4096,2048] -> xg:[2048,4096]
// 128 threads = 4 waves; each wave computes a 32x16 tile (2 M-subtiles).
// ---------------------------------------------------------------------------
__global__ __launch_bounds__(128) void xg_gemm_kernel(
    const unsigned short* __restrict__ xb,
    const unsigned short* __restrict__ Wib,
    const float* __restrict__ b_ih,
    float* __restrict__ xg)
{
  const int lane = threadIdx.x & 31;
  const int wave = threadIdx.x >> 5;
  const int n0 = blockIdx.x * 16;
  const int m0 = blockIdx.y * 128 + wave * 32;

  v8f acc0 = {}; v8f acc1 = {};
  for (int k = 0; k < 2048; k += 32) {
    v16bf a0 = load_a_frag(xb, 2048, m0,      k, lane);
    v16bf a1 = load_a_frag(xb, 2048, m0 + 16, k, lane);
    v16bf b  = load_bT_frag(Wib, 2048, n0,    k, lane);
    acc0 = WMMA_BF16(a0, b, acc0);
    acc1 = WMMA_BF16(a1, b, acc1);
  }
  const int   col   = n0 + (lane & 15);
  const float bias  = b_ih[col];
  const int   rbase = (lane & 16) ? 8 : 0;   // C/D layout: lanes16-31 hold M=8..15
#pragma unroll
  for (int r = 0; r < 8; ++r) {
    xg[(size_t)(m0 +      rbase + r) * 4096 + col] = acc0[r] + bias;
    xg[(size_t)(m0 + 16 + rbase + r) * 4096 + col] = acc1[r] + bias;
  }
}

// ---------------------------------------------------------------------------
// One LSTM step: gates = xg + h @ W_hh^T + b_hh, then c/h update.
// Each wave: 32x16 output tile, 4 gate accumulator chains (A-frag reuse x4).
// ---------------------------------------------------------------------------
__global__ __launch_bounds__(128) void lstm_step_kernel(
    const unsigned short* __restrict__ h_cur,   // [2048,1024] bf16
    const unsigned short* __restrict__ Whb,     // [4096,1024] bf16
    const float* __restrict__ xg,               // [2048,4096]
    const float* __restrict__ b_hh,             // [4096]
    float* __restrict__ c,                      // [2048,1024] (in-place)
    unsigned short* __restrict__ h_next,        // [2048,1024] bf16
    float* __restrict__ out_t)                  // [2048,1024] hs[t]
{
  const int lane = threadIdx.x & 31;
  const int wave = threadIdx.x >> 5;
  const int n0 = blockIdx.x * 16;
  const int m0 = blockIdx.y * 128 + wave * 32;

  v8f acc[2][4] = {};                           // [M-subtile][gate i,f,g,o]
  for (int k = 0; k < 1024; k += 32) {
    v16bf a0 = load_a_frag(h_cur, 1024, m0,      k, lane);
    v16bf a1 = load_a_frag(h_cur, 1024, m0 + 16, k, lane);
#pragma unroll
    for (int g = 0; g < 4; ++g) {
      v16bf b = load_bT_frag(Whb, 1024, g * 1024 + n0, k, lane);
      acc[0][g] = WMMA_BF16(a0, b, acc[0][g]);
      acc[1][g] = WMMA_BF16(a1, b, acc[1][g]);
    }
  }

  const int   col = n0 + (lane & 15);
  const float bi  = b_hh[col];
  const float bff = b_hh[col + 1024];
  const float bg  = b_hh[col + 2048];
  const float bo  = b_hh[col + 3072];
  const int   rbase = (lane & 16) ? 8 : 0;

#pragma unroll
  for (int s = 0; s < 2; ++s) {
#pragma unroll
    for (int r = 0; r < 8; ++r) {
      const int row = m0 + s * 16 + rbase + r;
      const float* xr = xg + (size_t)row * 4096 + col;
      const float iv = fast_sigmoid(acc[s][0][r] + xr[0]    + bi);
      const float fv = fast_sigmoid(acc[s][1][r] + xr[1024] + bff);
      const float gv = fast_tanh   (acc[s][2][r] + xr[2048] + bg);
      const float ov = fast_sigmoid(acc[s][3][r] + xr[3072] + bo);
      const size_t idx = (size_t)row * 1024 + col;
      const float cn = fv * c[idx] + iv * gv;
      c[idx] = cn;
      const float hn = ov * fast_tanh(cn);
      h_next[idx] = f32_to_bf16(hn);
      __builtin_nontemporal_store(hn, &out_t[idx]);   // hs never re-read: bypass L2
    }
  }
}

// ---------------------------------------------------------------------------
__global__ void cvt_f32_to_bf16_kernel(const float* __restrict__ src,
                                       unsigned short* __restrict__ dst, int n) {
  int i = blockIdx.x * blockDim.x + threadIdx.x;
  if (i < n) dst[i] = f32_to_bf16(src[i]);
}

__global__ void init_state_kernel(unsigned short* __restrict__ h,
                                  float* __restrict__ cc, int n) {
  int i = blockIdx.x * blockDim.x + threadIdx.x;
  if (i < n) { h[i] = f32_to_bf16(0.01f); cc[i] = 0.01f; }
}

// ---------------------------------------------------------------------------
extern "C" void kernel_launch(void* const* d_in, const int* in_sizes, int n_in,
                              void* d_out, int out_size, void* d_ws, size_t ws_size,
                              hipStream_t stream) {
  (void)in_sizes; (void)n_in; (void)out_size; (void)ws_size;
  const float* x    = (const float*)d_in[0];
  const float* W_ih = (const float*)d_in[1];
  const float* W_hh = (const float*)d_in[2];
  const float* b_ih = (const float*)d_in[3];
  const float* b_hh = (const float*)d_in[4];
  float* out = (float*)d_out;

  const int B = 2048, IN = 2048, H = 1024, FH = 4096, T = 15;

  char* w = (char*)d_ws;                             // ~84 MB workspace layout
  unsigned short* xb  = (unsigned short*)w;  w += (size_t)B  * IN * 2;   // 8 MB
  unsigned short* Wib = (unsigned short*)w;  w += (size_t)FH * IN * 2;   // 16 MB
  unsigned short* Whb = (unsigned short*)w;  w += (size_t)FH * H  * 2;   // 8 MB
  float*          xg  = (float*)w;           w += (size_t)B  * FH * 4;   // 32 MB
  float*          cst = (float*)w;           w += (size_t)B  * H  * 4;   // 8 MB
  unsigned short* h0  = (unsigned short*)w;  w += (size_t)B  * H  * 2;   // 4 MB
  unsigned short* h1  = (unsigned short*)w;                              // 4 MB

  int n;
  n = B * IN;  cvt_f32_to_bf16_kernel<<<(n + 255) / 256, 256, 0, stream>>>(x,    xb,  n);
  n = FH * IN; cvt_f32_to_bf16_kernel<<<(n + 255) / 256, 256, 0, stream>>>(W_ih, Wib, n);
  n = FH * H;  cvt_f32_to_bf16_kernel<<<(n + 255) / 256, 256, 0, stream>>>(W_hh, Whb, n);
  n = B * H;   init_state_kernel   <<<(n + 255) / 256, 256, 0, stream>>>(h0, cst, n);

  dim3 gridXG(FH / 16, B / 128);                     // 256 x 16 blocks, 128 thr
  xg_gemm_kernel<<<gridXG, 128, 0, stream>>>(xb, Wib, b_ih, xg);

  dim3 gridStep(H / 16, B / 128);                    // 64 x 16 blocks
  unsigned short* hc = h0; unsigned short* hn = h1;
  for (int t = 0; t < T; ++t) {
    lstm_step_kernel<<<gridStep, 128, 0, stream>>>(
        hc, Whb, xg, b_hh, cst, hn, out + (size_t)t * B * H);
    unsigned short* tmp = hc; hc = hn; hn = tmp;
  }
}
